// MiniMaxText01LinearAttention_24240795418685
// MI455X (gfx1250) — compile-verified, matrix operand order checked by reading, not run on hardware
//
#include <hip/hip_runtime.h>
#include <hip/hip_bf16.h>

typedef __attribute__((ext_vector_type(16))) _Float16 v16h;
typedef __attribute__((ext_vector_type(8)))  _Float16 v8h;
typedef __attribute__((ext_vector_type(4)))  _Float16 v4h;
typedef __attribute__((ext_vector_type(8)))  float    v8f;
typedef __attribute__((ext_vector_type(4)))  int      v4i;

#define H_    32
#define D_    64
#define HID_  2048
#define BLK_  256
#define N_    8192
#define NB_   (N_/BLK_)
#define QKVN_ (3*HID_)

#if defined(__has_builtin)
#if __has_builtin(__builtin_amdgcn_global_load_async_to_lds_b128)
#define HAVE_ASYNC_LDS 1
#endif
#endif

union FragA { v16h v; v8h h8[2]; _Float16 h[16]; };
union FragC { v8f  v; float f[8]; };

typedef __attribute__((address_space(1))) v4i* gptr_v4i;
typedef __attribute__((address_space(3))) v4i* lptr_v4i;

// async (or plain) 16-byte global -> LDS copy
__device__ __forceinline__ void cp16_g2l(const _Float16* g, _Float16* l) {
#ifdef HAVE_ASYNC_LDS
  __builtin_amdgcn_global_load_async_to_lds_b128((gptr_v4i)(g), (lptr_v4i)(l),
                                                 0, 0);
#else
  *(int4*)l = *(const int4*)g;
#endif
}
__device__ __forceinline__ void cp_wait() {
#ifdef HAVE_ASYNC_LDS
  asm volatile("s_wait_asynccnt 0" ::: "memory");
#endif
}

// 16-bit fragment: lane&15 -> row; per-lane data = two contiguous 8-half runs
// at K = hi*8 and K = hi*8+16.  Two ds_load_b128 per fragment.
__device__ __forceinline__ v16h frag_rowK(const _Float16* p, int ld) {
  const int lane = threadIdx.x & 31;
  const int row = lane & 15, hi = lane >> 4;
  const _Float16* q = p + row * ld + hi * 8;
  FragA fa;
  fa.h8[0] = *(const v8h*)(q);
  fa.h8[1] = *(const v8h*)(q + 16);
  return fa.v;
}

__device__ __forceinline__ v8f wmma_f16(v16h a, v16h b, v8f c) {
  return __builtin_amdgcn_wmma_f32_16x16x32_f16(false, a, false, b, (short)0, c,
                                                false, false);
}

// ---------------------------------------------------------------------------
// One-shot f32 -> f16 demotion (weights + activations), 8 elements/thread.
// ---------------------------------------------------------------------------
__global__ __launch_bounds__(256) void f32_to_f16_kernel(
    const float* __restrict__ s, _Float16* __restrict__ d, int n8) {
  const int i = blockIdx.x * 256 + threadIdx.x;
  if (i >= n8) return;
  const float4 a = ((const float4*)s)[2 * i];
  const float4 b = ((const float4*)s)[2 * i + 1];
  const v8h o = {(_Float16)a.x, (_Float16)a.y, (_Float16)a.z, (_Float16)a.w,
                 (_Float16)b.x, (_Float16)b.y, (_Float16)b.z, (_Float16)b.w};
  *(v8h*)(d + (size_t)i * 8) = o;
}

// ---------------------------------------------------------------------------
// WMMA GEMM (all-f16 inputs): C[m][n] = sum_k A[m][k] * W[n][k].
// Block tile 128x128, 8 waves (2 M x 4 N), wave tile 64x32, K-step 32.
// ---------------------------------------------------------------------------
template <int EPI>
__global__ __launch_bounds__(256) void gemm_wmma_kernel(
    const _Float16* __restrict__ A, const _Float16* __restrict__ W, int K,
    float* __restrict__ outF,
    _Float16* __restrict__ qh, _Float16* __restrict__ kh,
    _Float16* __restrict__ khT, _Float16* __restrict__ vhT,
    const float* __restrict__ hnorm, _Float16* __restrict__ X) {
  __shared__ _Float16 As[128][40];
  __shared__ _Float16 Bs[128][40];
  const int m0 = blockIdx.y * 128;
  const int n0 = blockIdx.x * 128;
  const int tid = threadIdx.x;
  const int wid = tid >> 5, lane = tid & 31;
  const int wm = (wid & 1) * 64;
  const int wn = (wid >> 1) * 32;

  const v8f z8 = {0.f, 0.f, 0.f, 0.f, 0.f, 0.f, 0.f, 0.f};
  FragC acc[4][2];
#pragma unroll
  for (int mt = 0; mt < 4; ++mt)
#pragma unroll
    for (int nt = 0; nt < 2; ++nt) acc[mt][nt].v = z8;

  for (int k0 = 0; k0 < K; k0 += 32) {
    // stage 128x32 f16 tiles: 512 b128 copies per side, 2 per thread
#pragma unroll
    for (int i = 0; i < 2; ++i) {
      const int lin = tid + i * 256;
      const int r = lin >> 2;          // 4 b128 per 32-half row
      const int c = (lin & 3) * 8;
      cp16_g2l(&A[(size_t)(m0 + r) * K + k0 + c], &As[r][c]);
      cp16_g2l(&W[(size_t)(n0 + r) * K + k0 + c], &Bs[r][c]);
    }
    cp_wait();
    __syncthreads();

    v16h af[4], bf[2];
#pragma unroll
    for (int mt = 0; mt < 4; ++mt) af[mt] = frag_rowK(&As[wm + 16 * mt][0], 40);
#pragma unroll
    for (int nt = 0; nt < 2; ++nt) bf[nt] = frag_rowK(&Bs[wn + 16 * nt][0], 40);
#pragma unroll
    for (int mt = 0; mt < 4; ++mt)
#pragma unroll
      for (int nt = 0; nt < 2; ++nt)
        acc[mt][nt].v = wmma_f16(af[mt], bf[nt], acc[mt][nt].v);
    __syncthreads();
  }

  const int rb = (lane >> 4) * 8, col = lane & 15;
#pragma unroll
  for (int mt = 0; mt < 4; ++mt)
#pragma unroll
    for (int nt = 0; nt < 2; ++nt)
#pragma unroll
      for (int r = 0; r < 8; ++r) {
        const int gm = m0 + wm + 16 * mt + rb + r;
        const int gn = n0 + wn + 16 * nt + col;
        const float v = acc[mt][nt].f[r];
        if constexpr (EPI == 0) {
          const float sv = v / (1.f + __expf(-v));  // silu
          const int hh = gn / 192;
          const int rem = gn - hh * 192;
          const int dd = rem & 63;
          const _Float16 hv = (_Float16)sv;
          if (rem < 64) {
            qh[((size_t)hh * N_ + gm) * D_ + dd] = hv;
          } else if (rem < 128) {
            kh[((size_t)hh * N_ + gm) * D_ + dd] = hv;
            khT[((size_t)hh * D_ + dd) * N_ + gm] = hv;
          } else {
            vhT[((size_t)hh * D_ + dd) * N_ + gm] = hv;
          }
        } else if constexpr (EPI == 1) {
          const float sg = 1.f / (1.f + __expf(-v));  // sigmoid gate
          X[(size_t)gm * HID_ + gn] =
              (_Float16)(sg * hnorm[(size_t)gm * HID_ + gn]);
        } else {
          outF[(size_t)gm * HID_ + gn] = v;
        }
      }
}

// ---------------------------------------------------------------------------
// Lightning attention: one workgroup per head, sequential scan over 32 blocks
// of 256 tokens.  Causal structure exploited: above-diagonal qk/intra chunks
// are skipped; strictly-causal chunks use factorized decay (no masking).
// ---------------------------------------------------------------------------
__global__ __launch_bounds__(256) void lightning_attn_kernel(
    const _Float16* __restrict__ qh, const _Float16* __restrict__ kh,
    const _Float16* __restrict__ khT, const _Float16* __restrict__ vhT,
    const float* __restrict__ kv_cache, float* __restrict__ hidden) {
  extern __shared__ char smem_raw[];
  _Float16* qs   = (_Float16*)smem_raw;        // [256][64]  row-major q
  _Float16* ks   = qs   + BLK_ * D_;           // [256][64]  row-major k
  _Float16* ksT  = ks   + BLK_ * D_;           // [64][256]  (k*k_decay)^T
  _Float16* vsT  = ksT  + D_ * BLK_;           // [64][256]  v^T
  _Float16* kvhT = vsT  + D_ * BLK_;           // [64][64]   kv^T f16 mirror
  _Float16* qkst = kvhT + D_ * D_;             // 8 waves x [32][32]
  float*    kvf  = (float*)(qkst + 8 * 32 * 32);  // [64][64] f32 kv state
  float*    qdec = kvf + D_ * D_;              // 256
  float*    ddec = qdec + BLK_;                // 256
  _Float16* kdech = (_Float16*)(ddec + BLK_);  // 256 f16 k-decay

  const int h = blockIdx.x;
  const int tid = threadIdx.x;
  const int wid = tid >> 5, lane = tid & 31;
  const int rb = (lane >> 4) * 8, col = lane & 15;
  // H=32 power of two: slope = 2^(-0.25*(h+1)) * (1 - 0/79 + 1e-5)
  const float s = exp2f(-0.25f * (float)(h + 1)) * 1.00001f;
  const float bdec = __expf(-s * 256.f);

  qdec[tid]  = __expf(-s * (float)(tid + 1));               // exp(-s*arr)
  ddec[tid]  = __expf(-s * (float)tid);                     // exp(-s*delta)
  kdech[tid] = (_Float16)__expf(-s * (float)(255 - tid));   // exp(-s*(B-arr))
  for (int i = tid; i < D_ * D_; i += 256)
    kvf[i] = kv_cache[(size_t)h * D_ * D_ + i];

  const _Float16* qg  = qh  + (size_t)h * N_ * D_;
  const _Float16* kg  = kh  + (size_t)h * N_ * D_;
  const _Float16* kTg = khT + (size_t)h * D_ * N_;
  const _Float16* vTg = vhT + (size_t)h * D_ * N_;
  const v8f z8 = {0.f, 0.f, 0.f, 0.f, 0.f, 0.f, 0.f, 0.f};
  __syncthreads();

  // lane-constant column decay factors: exp(-s*(31-cl)) for cl = 16*nt+col
  const float cfac[2] = {ddec[31 - col], ddec[15 - col]};

  for (int b = 0; b < NB_; ++b) {
    // ---- stage q,k row-major and v^T (async b128); ksT with decay fold ----
    const size_t base = (size_t)b * BLK_ * D_;
    for (int i = tid; i < BLK_ * D_ / 8; i += 256) {  // 2048 b128 each
      cp16_g2l(qg + base + i * 8, qs + i * 8);
      cp16_g2l(kg + base + i * 8, ks + i * 8);
    }
    for (int i = tid; i < D_ * BLK_ / 8; i += 256) {
      const int rowd = i >> 5;           // 32 b128 per 256-wide row
      const int c8 = (i & 31) * 8;
      cp16_g2l(vTg + (size_t)rowd * N_ + b * BLK_ + c8, vsT + rowd * BLK_ + c8);
      const v8h kv8 = *(const v8h*)(kTg + (size_t)rowd * N_ + b * BLK_ + c8);
      const v8h dv8 = *(const v8h*)(kdech + c8);
      *(v8h*)(ksT + rowd * BLK_ + c8) = kv8 * dv8;  // fold k_decay here
    }
    // transposed f16 mirror of kv state
    for (int i = tid; i < D_ * D_; i += 256)
      kvhT[i] = (_Float16)kvf[(i & 63) * D_ + (i >> 6)];
    cp_wait();
    __syncthreads();

    const int m0 = wid * 32;  // this wave's 32 rows of the block
    FragC accD[2][4];
#pragma unroll
    for (int mt = 0; mt < 2; ++mt)
#pragma unroll
      for (int nt = 0; nt < 4; ++nt) accD[mt][nt].v = z8;

    // ---- inter = (q * q_decay) @ kv,  B from kvhT ----
#pragma unroll
    for (int kk = 0; kk < 64; kk += 32) {
#pragma unroll
      for (int mt = 0; mt < 2; ++mt) {
        v16h a = frag_rowK(qs + (m0 + 16 * mt) * D_ + kk, D_);
        a = a * (_Float16)qdec[m0 + 16 * mt + (lane & 15)];
#pragma unroll
        for (int nt = 0; nt < 4; ++nt) {
          const v16h bf = frag_rowK(kvhT + 16 * nt * D_ + kk, D_);
          accD[mt][nt].v = wmma_f16(a, bf, accD[mt][nt].v);
        }
      }
    }

    _Float16* st = qkst + wid * 32 * 32;  // per-wave staging

    // ---- qk/intra only for causal chunks (nT <= wid) ----
    for (int nT = 0; nT <= wid; ++nT) {
      const int n0 = nT * 32;
      FragC qkacc[2][2];
#pragma unroll
      for (int mt = 0; mt < 2; ++mt)
#pragma unroll
        for (int nt = 0; nt < 2; ++nt) qkacc[mt][nt].v = z8;
#pragma unroll
      for (int kk = 0; kk < 64; kk += 32) {
        v16h am[2], bn[2];
#pragma unroll
        for (int mt = 0; mt < 2; ++mt)
          am[mt] = frag_rowK(qs + (m0 + 16 * mt) * D_ + kk, D_);
#pragma unroll
        for (int nt = 0; nt < 2; ++nt)
          bn[nt] = frag_rowK(ks + (n0 + 16 * nt) * D_ + kk, D_);
#pragma unroll
        for (int mt = 0; mt < 2; ++mt)
#pragma unroll
          for (int nt = 0; nt < 2; ++nt)
            qkacc[mt][nt].v = wmma_f16(am[mt], bn[nt], qkacc[mt][nt].v);
      }
      if (nT == wid) {
        // diagonal chunk: masked per-element decay
#pragma unroll
        for (int mt = 0; mt < 2; ++mt)
#pragma unroll
          for (int nt = 0; nt < 2; ++nt)
#pragma unroll
            for (int r = 0; r < 8; ++r) {
              const int rl = 16 * mt + rb + r;
              const int cl = 16 * nt + col;
              const int dm = rl - cl;  // m0 == n0 on the diagonal
              const float val =
                  (dm >= 0) ? qkacc[mt][nt].f[r] * ddec[dm] : 0.f;
              st[rl * 32 + cl] = (_Float16)val;
            }
      } else {
        // strictly causal: exp(-s(m-n)) = ddec[m-n0-31] * ddec[31-cl]
        const int basea = m0 - n0 - 31;
#pragma unroll
        for (int mt = 0; mt < 2; ++mt)
#pragma unroll
          for (int r = 0; r < 8; ++r) {
            const int rl = 16 * mt + rb + r;
            const float rf = ddec[basea + rl];
#pragma unroll
            for (int nt = 0; nt < 2; ++nt)
              st[rl * 32 + 16 * nt + col] =
                  (_Float16)(qkacc[mt][nt].f[r] * (rf * cfac[nt]));
          }
      }
      // intra += qk_decayed @ v
#pragma unroll
      for (int mt = 0; mt < 2; ++mt) {
        const v16h aq = frag_rowK(st + 16 * mt * 32, 32);
#pragma unroll
        for (int nt = 0; nt < 4; ++nt) {
          const v16h bv = frag_rowK(vsT + 16 * nt * BLK_ + n0, BLK_);
          accD[mt][nt].v = wmma_f16(aq, bv, accD[mt][nt].v);
        }
      }
    }

    // ---- kv partial over all chunks: decayed-k^T @ v ----
    FragC kvacc[2];
    kvacc[0].v = z8; kvacc[1].v = z8;
    const int t0 = wid * 2;
#pragma unroll
    for (int nT = 0; nT < 8; ++nT) {
      const int n0 = nT * 32;
#pragma unroll
      for (int ti = 0; ti < 2; ++ti) {
        const int t = t0 + ti;
        const int dr = (t >> 2) * 16, dc = (t & 3) * 16;
        const v16h ak = frag_rowK(ksT + dr * BLK_ + n0, BLK_);
        const v16h bv = frag_rowK(vsT + dc * BLK_ + n0, BLK_);
        kvacc[ti].v = wmma_f16(ak, bv, kvacc[ti].v);
      }
    }

    // ---- store attention output (inter + intra) ----
    float* hp = hidden + (size_t)(b * BLK_ + m0) * HID_ + h * D_;
#pragma unroll
    for (int mt = 0; mt < 2; ++mt)
#pragma unroll
      for (int nt = 0; nt < 4; ++nt)
#pragma unroll
        for (int r = 0; r < 8; ++r)
          hp[(16 * mt + rb + r) * HID_ + 16 * nt + col] = accD[mt][nt].f[r];

    __syncthreads();
    // ---- kv_new = block_decay * kv + sum((k*k_decay)^T @ v) ----
#pragma unroll
    for (int ti = 0; ti < 2; ++ti) {
      const int t = t0 + ti;
      const int dr = (t >> 2) * 16, dc = (t & 3) * 16;
#pragma unroll
      for (int r = 0; r < 8; ++r) {
        const int d = dr + rb + r, e = dc + col;
        kvf[d * D_ + e] = bdec * kvf[d * D_ + e] + kvacc[ti].f[r];
      }
    }
    __syncthreads();
  }
}

// ---------------------------------------------------------------------------
__global__ __launch_bounds__(256) void rmsnorm_kernel(
    const float* __restrict__ hidden, const float* __restrict__ w,
    float* __restrict__ out) {
  __shared__ float red[256];
  const int row = blockIdx.x;
  const int tid = threadIdx.x;
  const float* x = hidden + (size_t)row * HID_;
  float ss = 0.f;
  for (int i = tid; i < HID_; i += 256) { const float v = x[i]; ss += v * v; }
  red[tid] = ss;
  __syncthreads();
  for (int off = 128; off; off >>= 1) {
    if (tid < off) red[tid] += red[tid + off];
    __syncthreads();
  }
  const float inv = rsqrtf(red[0] / (float)HID_ + 1e-5f);
  for (int i = tid; i < HID_; i += 256)
    out[(size_t)row * HID_ + i] = x[i] * inv * w[i];
}

// ---------------------------------------------------------------------------
extern "C" void kernel_launch(void* const* d_in, const int* in_sizes, int n_in,
                              void* d_out, int out_size, void* d_ws,
                              size_t ws_size, hipStream_t stream) {
  (void)in_sizes; (void)n_in; (void)out_size; (void)ws_size;
  const float* hs   = (const float*)d_in[0];  // (8192, 2048)
  const float* kvc  = (const float*)d_in[1];  // (32, 64, 64)
  const float* Wqkv = (const float*)d_in[2];  // (6144, 2048)
  const float* Wg   = (const float*)d_in[3];  // (2048, 2048)
  const float* Wo   = (const float*)d_in[4];  // (2048, 2048)
  const float* nw   = (const float*)d_in[5];  // (2048,)
  float* out = (float*)d_out;

  // workspace layout (f16 first, then f32)
  _Float16* hsh   = (_Float16*)d_ws;                      // (N,HID)
  _Float16* Wqkvh = hsh   + (size_t)N_ * HID_;            // (6144,2048)
  _Float16* Wgh   = Wqkvh + (size_t)QKVN_ * HID_;         // (2048,2048)
  _Float16* Woh   = Wgh   + (size_t)HID_ * HID_;          // (2048,2048)
  _Float16* qh    = Woh   + (size_t)HID_ * HID_;          // (H,N,D)
  _Float16* kh    = qh    + (size_t)H_ * N_ * D_;         // (H,N,D)
  _Float16* khT   = kh    + (size_t)H_ * N_ * D_;         // (H,D,N)
  _Float16* vhT   = khT   + (size_t)H_ * N_ * D_;         // (H,D,N)
  _Float16* Xh    = vhT   + (size_t)H_ * N_ * D_;         // (N,HID) f16
  float* hidden   = (float*)(Xh + (size_t)N_ * HID_);     // (N,HID) f32
  float* hnorm    = hidden + (size_t)N_ * HID_;           // (N,HID) f32

  constexpr int ATTN_SMEM =
      (int)((2 * BLK_ * D_ + 2 * D_ * BLK_ + D_ * D_ + 8 * 32 * 32 + BLK_) *
                sizeof(short) +
            (D_ * D_ + 2 * BLK_) * sizeof(float));  // 174592 B
  (void)hipFuncSetAttribute(
      reinterpret_cast<const void*>(&lightning_attn_kernel),
      hipFuncAttributeMaxDynamicSharedMemorySize, ATTN_SMEM);

  const dim3 blk(256);
  // 0) demote activations + weights to f16 (one-shot)
  f32_to_f16_kernel<<<dim3(N_ * HID_ / 8 / 256), blk, 0, stream>>>(
      hs, hsh, N_ * HID_ / 8);
  f32_to_f16_kernel<<<dim3(QKVN_ * HID_ / 8 / 256), blk, 0, stream>>>(
      Wqkv, Wqkvh, QKVN_ * HID_ / 8);
  f32_to_f16_kernel<<<dim3(HID_ * HID_ / 8 / 256), blk, 0, stream>>>(
      Wg, Wgh, HID_ * HID_ / 8);
  f32_to_f16_kernel<<<dim3(HID_ * HID_ / 8 / 256), blk, 0, stream>>>(
      Wo, Woh, HID_ * HID_ / 8);
  // 1) qkv = silu(hs @ Wqkv^T) -> q,(k,k^T),v^T f16
  gemm_wmma_kernel<0><<<dim3(QKVN_ / 128, N_ / 128), blk, 0, stream>>>(
      hsh, Wqkvh, HID_, nullptr, qh, kh, khT, vhT, nullptr, nullptr);
  // 2) lightning attention scan, per head
  lightning_attn_kernel<<<dim3(H_), blk, ATTN_SMEM, stream>>>(
      qh, kh, khT, vhT, kvc, hidden);
  // 3) rms norm
  rmsnorm_kernel<<<dim3(N_), blk, 0, stream>>>(hidden, nw, hnorm);
  // 4) X = sigmoid(hs @ Wg^T) * hnorm  (f16)
  gemm_wmma_kernel<1><<<dim3(HID_ / 128, N_ / 128), blk, 0, stream>>>(
      hsh, Wgh, HID_, nullptr, nullptr, nullptr, nullptr, nullptr, hnorm, Xh);
  // 5) out = X @ Wo^T
  gemm_wmma_kernel<2><<<dim3(HID_ / 128, N_ / 128), blk, 0, stream>>>(
      Xh, Woh, HID_, out, nullptr, nullptr, nullptr, nullptr, nullptr,
      nullptr);
}
